// PolicyGCNLSTM_12266426598135
// MI455X (gfx1250) — matrix operand order repaired and checked by reading, not compiled, not run on hardware
//
#include <hip/hip_runtime.h>
#include <hip/hip_bf16.h>

typedef __attribute__((ext_vector_type(16))) _Float16 v16h;
typedef __attribute__((ext_vector_type(8)))  float    v8f;
typedef __attribute__((ext_vector_type(2)))  _Float16 h2;

#define NNODES 50000
#define NEDGES 800000
#define HDIM   64
#define LSEQ   2048

__device__ __forceinline__ float sigf(float x) { return 1.0f / (1.0f + __expf(-x)); }

__device__ __forceinline__ void atomic_add_f32(float* p, float v) {
  (void)__hip_atomic_fetch_add(p, v, __ATOMIC_RELAXED, __HIP_MEMORY_SCOPE_AGENT);
}

// ---------------------------------------------------------------------------
// Kernel 1: xw = x @ W1   ([N,64] x [64,64]) via v_wmma_f32_16x16x32_f16.
// Block: 256 threads = 8 waves; tile 32 rows x 64 cols; each wave one 16x16
// C tile; K=64 as two K32 WMMA steps. B stored transposed [n][k] in LDS so
// each lane's 16 K-halves are contiguous (2x ds_load_b128 per fragment).
// ---------------------------------------------------------------------------
__global__ void __launch_bounds__(256)
gemm_xw_kernel(const float* __restrict__ x, const float* __restrict__ W,
               float* __restrict__ xw, int nrows) {
  __shared__ _Float16 Ah[32 * 64];    // [row][k]  4 KB
  __shared__ _Float16 BhT[64 * 64];   // [n][k]    8 KB
  const int tid  = threadIdx.x;
  const int row0 = blockIdx.x * 32;

  for (int i = tid; i < 64 * 64; i += 256) {
    int n = i >> 6, k = i & 63;
    BhT[i] = (_Float16)W[k * 64 + n];          // W1 is [k][n]
  }
  for (int i = tid; i < 32 * 64; i += 256) {
    int r = i >> 6, k = i & 63;
    int gr = row0 + r;
    Ah[i] = (_Float16)((gr < nrows) ? x[gr * 64 + k] : 0.0f);
  }
  __syncthreads();

  const int wave = tid >> 5;
  const int lane = tid & 31;
  const int tm   = wave >> 2;          // 0..1 row tile
  const int n0   = (wave & 3) * 16;    // col tile
  const int lh   = lane >> 4;          // half-wave select
  const int mrow = (lane & 15) + tm * 16;
  const int ncol = n0 + (lane & 15);

  v8f c = {};
  #pragma unroll
  for (int kk = 0; kk < 64; kk += 32) {
    v16h a, b;
    const int ka = kk + lh * 16;       // A: lanes 0-15 K 0-15; lanes 16-31 K 16-31
    #pragma unroll
    for (int e = 0; e < 16; ++e) a[e] = Ah[mrow * 64 + ka + e];
    const int kb = kk + lh * 16;       // B: lanes 0-15 K 0-15; lanes 16-31 K 16-31
    #pragma unroll
    for (int e = 0; e < 16; ++e) b[e] = BhT[ncol * 64 + kb + e];   // contiguous
    c = __builtin_amdgcn_wmma_f32_16x16x32_f16(false, a, false, b, (short)0, c, false, false);
  }

  const int mb = tm * 16 + lh * 8;     // C: lanes 16-31 hold M = r+8
  if (row0 + 32 <= nrows) {            // fast path: 1562 of 1563 blocks
    #pragma unroll
    for (int r = 0; r < 8; ++r) xw[(row0 + mb + r) * 64 + ncol] = c[r];
  } else {
    #pragma unroll
    for (int r = 0; r < 8; ++r) {
      int gr = row0 + mb + r;
      if (gr < nrows) xw[gr * 64 + ncol] = c[r];
    }
  }
}

// ---------------------------------------------------------------------------
// Init: deg = 1 (self loop), mask = 0, reduction cells = 0.
// ---------------------------------------------------------------------------
__global__ void __launch_bounds__(256)
init_kernel(float* deg, int* mask, unsigned long long* red, float* redsum, int n) {
  int i = blockIdx.x * blockDim.x + threadIdx.x;
  if (i < n) { deg[i] = 1.0f; mask[i] = 0; }
  if (i == 0) { *red = 0ULL; *redsum = 0.0f; }
}

__global__ void __launch_bounds__(256)
deg_kernel(const int* __restrict__ ei, const float* __restrict__ ew, float* deg) {
  int e = blockIdx.x * blockDim.x + threadIdx.x;
  if (e >= NEDGES) return;
  atomic_add_f32(&deg[ei[NEDGES + e]], ew[e]);
}

__global__ void __launch_bounds__(256)
rsqrt_kernel(float* deg, int n) {
  int i = blockIdx.x * blockDim.x + threadIdx.x;
  if (i < n) deg[i] = __frsqrt_rn(deg[i]);   // deg >= 1 always (self loop)
}

// z[n,h] = dinv[n]^2 * xw[n,h] + b1[h]   (self-loop message + bias)
__global__ void __launch_bounds__(256)
zinit_kernel(const float* __restrict__ xw, const float* __restrict__ dinv,
             const float* __restrict__ b1, float* __restrict__ z) {
  int i = blockIdx.x * blockDim.x + threadIdx.x;
  if (i >= NNODES * HDIM) return;
  int n = i >> 6, h = i & 63;
  float d = dinv[n];
  z[i] = d * d * xw[i] + b1[h];
}

// z[dst] += dinv[src]*ew*dinv[dst] * xw[src]   (0.4 GB random traffic; HBM-bound)
__global__ void __launch_bounds__(256)
scatter_kernel(const int* __restrict__ ei, const float* __restrict__ ew,
               const float* __restrict__ dinv, const float* __restrict__ xw,
               float* __restrict__ z) {
  int gid = blockIdx.x * blockDim.x + threadIdx.x;
  int e = gid >> 6, h = gid & 63;
  if (e >= NEDGES) return;
  int s = ei[e], d = ei[NEDGES + e];
  float nr = dinv[s] * ew[e] * dinv[d];
  atomic_add_f32(&z[d * 64 + h], nr * xw[s * 64 + h]);
}

// ---------------------------------------------------------------------------
// Kernel: Xproj[t,g] = z[visited[t]] @ Wih0^T + (bih0+bhh0)   [2048,256] WMMA.
// Block: 512 threads = 16 waves, tile 16 rows x 256 cols.
// A-tile gather staged with GLOBAL_LOAD_ASYNC_TO_LDS_B64 (ASYNCcnt path),
// B stored in Wih0-native [g][k] layout -> contiguous lane fragments.
// ---------------------------------------------------------------------------
__global__ void __launch_bounds__(512)
gemm_xproj_kernel(const float* __restrict__ z, const int* __restrict__ visited,
                  const float* __restrict__ Wih0, const float* __restrict__ bih0,
                  const float* __restrict__ bhh0, float* __restrict__ Xp) {
  __shared__ float    Az[16 * 64];    // fp32 async staging  4 KB
  __shared__ _Float16 Ah[16 * 64];    // [t][k]   2 KB
  __shared__ _Float16 BhT[64 * 256];  // [g][k]  32 KB (== Wih0 layout)
  const int tid = threadIdx.x;
  const int t0  = blockIdx.x * 16;

  // async gather: 512 lanes x 8 bytes == 16 rows x 64 floats
  {
    int flat = tid * 2;                      // first float of this lane's pair
    int r = flat >> 6, k = flat & 63;
    int node = visited[t0 + r];
    unsigned long long gaddr = (unsigned long long)(const void*)(z + (size_t)node * 64 + k);
    unsigned ldsa = (unsigned)(unsigned long long)(const void*)&Az[flat];
    asm volatile("global_load_async_to_lds_b64 %0, %1, off"
                 :: "v"(ldsa), "v"(gaddr) : "memory");
  }
  for (int i = tid; i < 64 * 256; i += 512)
    BhT[i] = (_Float16)Wih0[i];              // B[g][k] = Wih0[g][k] verbatim
  asm volatile("s_wait_asynccnt 0" ::: "memory");
  __syncthreads();
  for (int i = tid; i < 16 * 64; i += 512)
    Ah[i] = (_Float16)Az[i];
  __syncthreads();

  const int wave = tid >> 5;
  const int lane = tid & 31;
  const int n0   = wave * 16;
  const int lh   = lane >> 4;
  const int mrow = lane & 15;
  const int gcol = n0 + (lane & 15);

  v8f c = {};
  #pragma unroll
  for (int kk = 0; kk < 64; kk += 32) {
    v16h a, b;
    const int ka = kk + lh * 16;
    #pragma unroll
    for (int e = 0; e < 16; ++e) a[e] = Ah[mrow * 64 + ka + e];
    const int kb = kk + lh * 16;
    #pragma unroll
    for (int e = 0; e < 16; ++e) b[e] = BhT[gcol * 64 + kb + e];   // contiguous
    c = __builtin_amdgcn_wmma_f32_16x16x32_f16(false, a, false, b, (short)0, c, false, false);
  }

  float bias = bih0[gcol] + bhh0[gcol];
  #pragma unroll
  for (int r = 0; r < 8; ++r) {
    int t = t0 + lh * 8 + r;
    Xp[t * 256 + gcol] = c[r] + bias;
  }
}

// ---------------------------------------------------------------------------
// Recurrent LSTM (both layers fused). 1 block x 256 threads on one WGP.
// Weights packed as f16x2 [k/2][g] in LDS: one ds_load_b32 feeds two FMAs
// on the latency-critical serial path. ~100 KB dynamic LDS.
// ---------------------------------------------------------------------------
__global__ void __launch_bounds__(256)
lstm_kernel(const float* __restrict__ Xp, const float* __restrict__ Whh0,
            const float* __restrict__ Wih1, const float* __restrict__ Whh1,
            const float* __restrict__ bih1, const float* __restrict__ bhh1,
            float* __restrict__ finalh) {
  extern __shared__ char smraw[];
  h2* W0p  = (h2*)smraw;                     // [32][256] pairs
  h2* W1ip = W0p  + 32 * 256;
  h2* W1hp = W1ip + 32 * 256;
  float* fbase = (float*)(W1hp + 32 * 256);
  float* h0 = fbase;        float* c0 = h0 + 64;
  float* h1 = c0 + 64;      float* c1 = h1 + 64;
  float* g0 = c1 + 64;      // [256]
  float* g1 = g0 + 256;     // [256]
  float* b1s = g1 + 256;    // [256]

  const int tid = threadIdx.x;
  for (int i = tid; i < 32 * 256; i += 256) {
    int k2 = i >> 8, g = i & 255;
    W0p[i]  = h2{(_Float16)Whh0[g * 64 + 2 * k2], (_Float16)Whh0[g * 64 + 2 * k2 + 1]};
    W1ip[i] = h2{(_Float16)Wih1[g * 64 + 2 * k2], (_Float16)Wih1[g * 64 + 2 * k2 + 1]};
    W1hp[i] = h2{(_Float16)Whh1[g * 64 + 2 * k2], (_Float16)Whh1[g * 64 + 2 * k2 + 1]};
  }
  b1s[tid] = bih1[tid] + bhh1[tid];
  if (tid < 64) { h0[tid] = 0.f; c0[tid] = 0.f; h1[tid] = 0.f; c1[tid] = 0.f; }
  __syncthreads();

  for (int t = 0; t < LSEQ; ++t) {
    // layer 0 gate: precomputed x-proj + h0 @ Whh0^T
    float acc = Xp[t * 256 + tid];
    #pragma unroll 8
    for (int k2 = 0; k2 < 32; ++k2) {
      h2 w = W0p[k2 * 256 + tid];
      acc += h0[2 * k2] * (float)w[0] + h0[2 * k2 + 1] * (float)w[1];
    }
    g0[tid] = acc;
    __syncthreads();
    if (tid < 64) {
      float i_ = sigf(g0[tid]);
      float f_ = sigf(g0[64 + tid]);
      float gg = tanhf(g0[128 + tid]);
      float o_ = sigf(g0[192 + tid]);
      float c = f_ * c0[tid] + i_ * gg;
      c0[tid] = c;
      h0[tid] = o_ * tanhf(c);
    }
    __syncthreads();
    // layer 1 gate: h0_t @ Wih1^T + h1_{t-1} @ Whh1^T + bias
    float acc1 = b1s[tid];
    #pragma unroll 8
    for (int k2 = 0; k2 < 32; ++k2) {
      h2 wi = W1ip[k2 * 256 + tid];
      h2 wh = W1hp[k2 * 256 + tid];
      acc1 += h0[2 * k2] * (float)wi[0] + h0[2 * k2 + 1] * (float)wi[1];
      acc1 += h1[2 * k2] * (float)wh[0] + h1[2 * k2 + 1] * (float)wh[1];
    }
    g1[tid] = acc1;
    __syncthreads();
    if (tid < 64) {
      float i_ = sigf(g1[tid]);
      float f_ = sigf(g1[64 + tid]);
      float gg = tanhf(g1[128 + tid]);
      float o_ = sigf(g1[192 + tid]);
      float c = f_ * c1[tid] + i_ * gg;
      c1[tid] = c;
      h1[tid] = o_ * tanhf(c);
    }
    __syncthreads();
  }
  if (tid < 64) finalh[tid] = h1[tid];
}

// ---------------------------------------------------------------------------
// scores[n] = z[n] . final_hidden  (wave per row, shuffle reduce)
// ---------------------------------------------------------------------------
__global__ void __launch_bounds__(256)
scores_kernel(const float* __restrict__ z, const float* __restrict__ fh,
              float* __restrict__ scores, int n) {
  int gwave = (blockIdx.x * blockDim.x + threadIdx.x) >> 5;
  int lane  = threadIdx.x & 31;
  if (gwave >= n) return;
  float v = z[gwave * 64 + lane] * fh[lane] + z[gwave * 64 + 32 + lane] * fh[32 + lane];
  #pragma unroll
  for (int o = 16; o > 0; o >>= 1) v += __shfl_xor(v, o, 32);
  if (lane == 0) scores[gwave] = v;
}

__global__ void __launch_bounds__(256)
mark_kernel(const int* __restrict__ visited, int* __restrict__ mask) {
  int t = blockIdx.x * blockDim.x + threadIdx.x;
  if (t < LSEQ) mask[visited[t]] = 1;
}

// packed (orderable-key<<32 | index) atomicMax -> max AND argmax in one pass
__global__ void __launch_bounds__(256)
redmax_kernel(const float* __restrict__ scores, const int* __restrict__ mask,
              unsigned long long* red, int n) {
  int i = blockIdx.x * blockDim.x + threadIdx.x;
  if (i >= n || mask[i]) return;
  unsigned u   = __float_as_uint(scores[i]);
  unsigned key = (u & 0x80000000u) ? ~u : (u | 0x80000000u);
  unsigned long long p = ((unsigned long long)key << 32) | (unsigned)i;
  (void)__hip_atomic_fetch_max(red, p, __ATOMIC_RELAXED, __HIP_MEMORY_SCOPE_AGENT);
}

__global__ void __launch_bounds__(256)
exp_kernel(const float* __restrict__ scores, const int* __restrict__ mask,
           const unsigned long long* __restrict__ red, float* __restrict__ probs,
           float* redsum, int n) {
  int i = blockIdx.x * blockDim.x + threadIdx.x;
  unsigned key = (unsigned)(*red >> 32);
  unsigned u   = (key & 0x80000000u) ? (key & 0x7fffffffu) : ~key;
  float mx = __uint_as_float(u);
  float e = 0.0f;
  if (i < n) {
    e = mask[i] ? 0.0f : __expf(scores[i] - mx);
    probs[i] = e;
  }
  float v = e;
  #pragma unroll
  for (int o = 16; o > 0; o >>= 1) v += __shfl_xor(v, o, 32);
  if ((threadIdx.x & 31) == 0) atomic_add_f32(redsum, v);
}

__global__ void __launch_bounds__(256)
norm_kernel(float* __restrict__ probs, const float* __restrict__ redsum,
            const unsigned long long* __restrict__ red, float* __restrict__ out0, int n) {
  int i = blockIdx.x * blockDim.x + threadIdx.x;
  float inv = 1.0f / (*redsum);
  if (i < n) probs[i] *= inv;
  if (i == 0) *out0 = (float)(unsigned)(*red & 0xffffffffu);  // argmax as next_node
}

// ---------------------------------------------------------------------------
extern "C" void kernel_launch(void* const* d_in, const int* in_sizes, int n_in,
                              void* d_out, int out_size, void* d_ws, size_t ws_size,
                              hipStream_t stream) {
  const float* x       = (const float*)d_in[0];
  const int*   ei      = (const int*)d_in[1];
  const float* ew      = (const float*)d_in[2];
  const int*   visited = (const int*)d_in[3];
  const float* W1      = (const float*)d_in[4];
  const float* b1      = (const float*)d_in[5];
  const float* Wih0    = (const float*)d_in[6];
  const float* Whh0    = (const float*)d_in[7];
  const float* bih0    = (const float*)d_in[8];
  const float* bhh0    = (const float*)d_in[9];
  const float* Wih1    = (const float*)d_in[10];
  const float* Whh1    = (const float*)d_in[11];
  const float* bih1    = (const float*)d_in[12];
  const float* bhh1    = (const float*)d_in[13];
  float* out = (float*)d_out;

  // workspace carve (~28.3 MB)
  float* xw     = (float*)d_ws;
  float* z      = xw + (size_t)NNODES * HDIM;
  float* deg    = z + (size_t)NNODES * HDIM;      // becomes dinv in place
  float* Xp     = deg + NNODES;
  float* finalh = Xp + (size_t)LSEQ * 256;
  float* scores = finalh + 64;
  int*   mask   = (int*)(scores + NNODES);
  unsigned long long* red = (unsigned long long*)(mask + NNODES);
  float* redsum = (float*)(red + 1);

  gemm_xw_kernel<<<(NNODES + 31) / 32, 256, 0, stream>>>(x, W1, xw, NNODES);
  init_kernel<<<(NNODES + 255) / 256, 256, 0, stream>>>(deg, mask, red, redsum, NNODES);
  deg_kernel<<<(NEDGES + 255) / 256, 256, 0, stream>>>(ei, ew, deg);
  rsqrt_kernel<<<(NNODES + 255) / 256, 256, 0, stream>>>(deg, NNODES);
  zinit_kernel<<<(NNODES * HDIM + 255) / 256, 256, 0, stream>>>(xw, deg, b1, z);
  scatter_kernel<<<(NEDGES * HDIM) / 256, 256, 0, stream>>>(ei, ew, deg, xw, z);
  gemm_xproj_kernel<<<LSEQ / 16, 512, 0, stream>>>(z, visited, Wih0, bih0, bhh0, Xp);

  size_t smem = 3u * 32 * 256 * sizeof(h2) + (4 * 64 + 3 * 256) * sizeof(float);
  lstm_kernel<<<1, 256, smem, stream>>>(Xp, Whh0, Wih1, Whh1, bih1, bhh1, finalh);

  scores_kernel<<<(NNODES * 32 + 255) / 256, 256, 0, stream>>>(z, finalh, scores, NNODES);
  mark_kernel<<<(LSEQ + 255) / 256, 256, 0, stream>>>(visited, mask);
  redmax_kernel<<<(NNODES + 255) / 256, 256, 0, stream>>>(scores, mask, red, NNODES);
  exp_kernel<<<(NNODES + 255) / 256, 256, 0, stream>>>(scores, mask, red, out + 1, redsum, NNODES);
  norm_kernel<<<(NNODES + 255) / 256, 256, 0, stream>>>(out + 1, redsum, red, out, NNODES);
}
